// DynamicConvolution_23347442221639
// MI455X (gfx1250) — matrix operand order repaired and checked by reading, not compiled, not run on hardware
//
#include <hip/hip_runtime.h>
#include <hip/hip_bf16.h>

// DynamicConvolution for MI455X (gfx1250), wave32 + BF16 WMMA path.
// [K0]  pack W1/Ww/W2 -> bf16 WMMA fragments (one-time, 2MB, L2-resident)
// [K0q] convert q -> bf16 (so all GEMM A-feeds are pure b128 loads)
// [K1]  q@W1+GLU -> x(bf16)      (16x64 per wave, 8 accums)
// [K2]  x@Ww+bias -> banded softmax -> w(f32)  (A stripe register-resident)
// [K3]  depthwise dynamic conv + mask -> y(bf16)
// [K4]  y@W2+b2 -> out(f32)      (16x64 per wave, 4 accums)

namespace {

constexpr int Bb   = 8;
constexpr int T    = 4096;
constexpr int C    = 512;
constexpr int H    = 16;
constexpr int KW   = 31;
constexpr int PADW = 15;
constexpr int M    = Bb * T;     // 32768 rows
constexpr int HK   = H * KW;     // 496
constexpr int C2   = 2 * C;      // 1024

typedef __bf16 bf16_t;
typedef __attribute__((ext_vector_type(16))) __bf16 v16bf;
typedef __attribute__((ext_vector_type(8)))  __bf16 v8bf;
typedef __attribute__((ext_vector_type(8)))  float  v8f;

__device__ __forceinline__ v8f wmma_bf16(v16bf a, v16bf b, v8f c) {
  return __builtin_amdgcn_wmma_f32_16x16x32_bf16(false, a, false, b, (short)0, c, false, false);
}

// ---- WMMA operand loaders (layouts per CDNA5 ISA §7.12.2) ----

// A-matrix 16x32 bf16, row-major bf16 source: two 16B vector loads.
// lanes 0-15: row=lane,    elems 0..7 = K k0+0..7,  elems 8..15 = K k0+16..23
// lanes16-31: row=lane-16, elems 0..7 = K k0+8..15, elems 8..15 = K k0+24..31
__device__ __forceinline__ v16bf load_a_bf16(const bf16_t* __restrict__ X, int ld,
                                             int row0, int k0, int lane) {
  const int row = row0 + (lane & 15);
  const int kb  = (lane < 16) ? 0 : 8;
  const bf16_t* p = X + (size_t)row * ld + k0 + kb;
  v8bf lo = *(const v8bf*)p;
  v8bf hi = *(const v8bf*)(p + 16);
  return __builtin_shufflevector(lo, hi, 0, 1, 2, 3, 4, 5, 6, 7,
                                 8, 9, 10, 11, 12, 13, 14, 15);
}

// B fragment: pre-packed 32x16 (KxN) tile, 32 lanes x 16 bf16 contiguous.
__device__ __forceinline__ v16bf load_b_frag(const bf16_t* __restrict__ Wf, int nTiles,
                                             int bk, int bn, int lane) {
  return *(const v16bf*)(Wf + (((size_t)bk * nTiles + bn) * 32 + lane) * 16);
}

// ---- K0: pack an fp32 weight matrix (C x N, row-major) into bf16 fragments ----
__global__ void k0_pack_b(const float* __restrict__ W, bf16_t* __restrict__ Wf, int N) {
  const int lane = threadIdx.x & 31;
  const int frag = blockIdx.x * 8 + (threadIdx.x >> 5);
  const int nTiles = N >> 4;
  const int bk = frag / nTiles;
  const int bn = frag - bk * nTiles;
  const int col = bn * 16 + (lane & 15);
  const int krow = bk * 32 + ((lane < 16) ? 0 : 8);
  const float* src = W + (size_t)krow * N + col;
  bf16_t* dst = Wf + ((size_t)frag * 32 + lane) * 16;
#pragma unroll
  for (int e = 0; e < 16; ++e) dst[e] = (bf16_t)src[(size_t)(e < 8 ? e : e + 8) * N];
}

// ---- K0q: elementwise fp32 -> bf16 convert (8 elems/thread, vector I/O) ----
__global__ void k0_cvt_bf16(const float* __restrict__ S, bf16_t* __restrict__ D) {
  const size_t i = ((size_t)blockIdx.x * blockDim.x + threadIdx.x) * 8;
#pragma unroll
  for (int e = 0; e < 8; ++e) D[i + e] = (bf16_t)S[i + e];
}

// ---- K1: h = q@W1 + b1, GLU -> x (bf16). 16 rows x 64 cols per wave. ----
__global__ void k1_gemm_glu(const bf16_t* __restrict__ Qb, const bf16_t* __restrict__ W1f,
                            const float* __restrict__ b1, bf16_t* __restrict__ Xb) {
  const int lane  = threadIdx.x & 31;
  const int wave  = threadIdx.x >> 5;
  const int tile  = blockIdx.x * 8 + wave;   // M/16 * C/64 = 16384 tiles
  const int mTile = tile >> 3;
  const int ng    = tile & 7;                // 64-col group of x
  const int row0  = mTile * 16;
  v8f accA[4] = {{}, {}, {}, {}};
  v8f accG[4] = {{}, {}, {}, {}};
  for (int k0 = 0; k0 < C; k0 += 32) {
    const int bk = k0 >> 5;
    const v16bf a = load_a_bf16(Qb, C, row0, k0, lane);
#pragma unroll
    for (int u = 0; u < 4; ++u) {
      const v16bf bA = load_b_frag(W1f, 64, bk, ng * 4 + u,      lane);  // a-half cols
      const v16bf bG = load_b_frag(W1f, 64, bk, 32 + ng * 4 + u, lane);  // g-half (+C)
      accA[u] = wmma_bf16(a, bA, accA[u]);
      accG[u] = wmma_bf16(a, bG, accG[u]);
    }
  }
  const int rbase = row0 + ((lane < 16) ? 0 : 8);
#pragma unroll
  for (int u = 0; u < 4; ++u) {
    const int colx = ng * 64 + u * 16 + (lane & 15);
    const float ba = b1[colx];
    const float bg = b1[C + colx];
#pragma unroll
    for (int i = 0; i < 8; ++i) {
      const float av = accA[u][i] + ba;
      const float gv = accG[u][i] + bg;
      const float xv = av / (1.0f + __expf(-gv));   // a * sigmoid(g)
      Xb[(size_t)(rbase + i) * C + colx] = (bf16_t)xv;
    }
  }
}

// ---- K2: logits = x@Ww + bw (into LDS), banded softmax over K=31 -> Wsm (f32) ----
__global__ void k2_weights_softmax(const bf16_t* __restrict__ Xb, const bf16_t* __restrict__ Wwf,
                                   const float* __restrict__ bw, float* __restrict__ Wsm) {
  __shared__ float lds[16 * HK];               // 31744 B
  const int lane = threadIdx.x & 31;
  const int wave = threadIdx.x >> 5;
  const int row0 = blockIdx.x * 16;
  // whole 16x512 A stripe register-resident: 16 fragments x 8 VGPRs
  v16bf afr[16];
#pragma unroll
  for (int bk = 0; bk < 16; ++bk) afr[bk] = load_a_bf16(Xb, C, row0, bk * 32, lane);
  for (int nt = wave; nt < 31; nt += 8) {      // 31 N-tiles over 8 waves (uniform per wave)
    v8f acc = {};
#pragma unroll
    for (int bk = 0; bk < 16; ++bk)
      acc = wmma_bf16(afr[bk], load_b_frag(Wwf, 31, bk, nt, lane), acc);
    const int colx = nt * 16 + (lane & 15);
    const int rb   = (lane < 16) ? 0 : 8;
    const float bias = bw[colx];
#pragma unroll
    for (int i = 0; i < 8; ++i) lds[(rb + i) * HK + colx] = acc[i] + bias;
  }
  __syncthreads();
  // banded softmax: one thread per (row-in-tile, head); output t attends c=t+j-PAD
  const int r = threadIdx.x >> 4;
  const int h = threadIdx.x & 15;
  const int m = row0 + r;
  const int t = m & (T - 1);
  const float* src = &lds[r * HK + h * KW];
  float vals[KW];
  float mx = -3.0e38f;
#pragma unroll
  for (int j = 0; j < KW; ++j) {
    const int cc = t + j - PADW;
    const float v = (cc >= 0 && cc < T) ? src[j] : -INFINITY;
    vals[j] = v;
    mx = fmaxf(mx, v);
  }
  float s = 0.f;
#pragma unroll
  for (int j = 0; j < KW; ++j) { const float e = __expf(vals[j] - mx); vals[j] = e; s += e; }
  const float inv = 1.f / s;
  float* dst = Wsm + (size_t)m * HK + h * KW;
#pragma unroll
  for (int j = 0; j < KW; ++j) dst[j] = vals[j] * inv;
}

// ---- K3: depthwise dynamic conv + frame mask -> y (bf16) ----
__global__ void k3_dynconv(const bf16_t* __restrict__ Xb, const float* __restrict__ Wsm,
                           const int* __restrict__ mask, bf16_t* __restrict__ Yb) {
  const int idx = blockIdx.x * blockDim.x + threadIdx.x;  // exactly M*C threads
  const int m = idx >> 9;          // / C
  const int c = idx & (C - 1);
  const int h = c >> 5;            // / D
  const int t = m & (T - 1);
  const float* wp = Wsm + (size_t)m * HK + h * KW;
  float acc = 0.f;
#pragma unroll
  for (int j = 0; j < KW; ++j) {
    const int tt = t + j - PADW;
    if (tt >= 0 && tt < T)
      acc += wp[j] * (float)Xb[(size_t)(m + j - PADW) * C + c];
  }
  if (mask[m] == 0) acc = 0.f;     // mask is (B,1,T) -> flat index b*T+t == m
  Yb[(size_t)m * C + c] = (bf16_t)acc;
}

// ---- K4: out = y@W2 + b2 (f32). 16 rows x 64 cols per wave. ----
__global__ void k4_gemm_out(const bf16_t* __restrict__ Yb, const bf16_t* __restrict__ W2f,
                            const float* __restrict__ b2, float* __restrict__ Out) {
  const int lane  = threadIdx.x & 31;
  const int wave  = threadIdx.x >> 5;
  const int tile  = blockIdx.x * 8 + wave;   // M/16 * C/64 = 16384 tiles
  const int mTile = tile >> 3;
  const int ng    = tile & 7;
  const int row0  = mTile * 16;
  v8f acc[4] = {{}, {}, {}, {}};
  for (int k0 = 0; k0 < C; k0 += 32) {
    const int bk = k0 >> 5;
    const v16bf a = load_a_bf16(Yb, C, row0, k0, lane);
#pragma unroll
    for (int u = 0; u < 4; ++u) {
      const v16bf b = load_b_frag(W2f, 32, bk, ng * 4 + u, lane);
      acc[u] = wmma_bf16(a, b, acc[u]);
    }
  }
  const int rbase = row0 + ((lane < 16) ? 0 : 8);
#pragma unroll
  for (int u = 0; u < 4; ++u) {
    const int colx = ng * 64 + u * 16 + (lane & 15);
    const float bias = b2[colx];
#pragma unroll
    for (int i = 0; i < 8; ++i)
      Out[(size_t)(rbase + i) * C + colx] = acc[u][i] + bias;
  }
}

} // namespace

extern "C" void kernel_launch(void* const* d_in, const int* in_sizes, int n_in,
                              void* d_out, int out_size, void* d_ws, size_t ws_size,
                              hipStream_t stream) {
  (void)in_sizes; (void)n_in; (void)out_size; (void)ws_size;
  const float* Q    = (const float*)d_in[0];
  // d_in[1]=key, d_in[2]=value: unused by the reference
  const int*   mask = (const int*)  d_in[3];
  const float* W1   = (const float*)d_in[4];
  const float* b1   = (const float*)d_in[5];
  const float* Ww   = (const float*)d_in[6];
  const float* bw   = (const float*)d_in[7];
  const float* W2   = (const float*)d_in[8];
  const float* b2   = (const float*)d_in[9];
  float* Out = (float*)d_out;

  // Workspace: Qb(32MB) | Xb(32MB) | Yb(32MB) | Wsm(62MB) | packed weights(~2MB)
  char* ws = (char*)d_ws;
  size_t off = 0;
  bf16_t* Qb  = (bf16_t*)(ws + off); off += (size_t)M * C * sizeof(bf16_t);
  bf16_t* Xb  = (bf16_t*)(ws + off); off += (size_t)M * C * sizeof(bf16_t);
  bf16_t* Yb  = (bf16_t*)(ws + off); off += (size_t)M * C * sizeof(bf16_t);
  float*  Wsm = (float*) (ws + off); off += (size_t)M * HK * sizeof(float);
  bf16_t* W1f = (bf16_t*)(ws + off); off += (size_t)(C / 32) * (C2 / 16) * 512 * sizeof(bf16_t);
  bf16_t* Wwf = (bf16_t*)(ws + off); off += (size_t)(C / 32) * (HK / 16) * 512 * sizeof(bf16_t);
  bf16_t* W2f = (bf16_t*)(ws + off); off += (size_t)(C / 32) * (C  / 16) * 512 * sizeof(bf16_t);

  // K0: pack weights to bf16 fragments; convert q to bf16
  k0_pack_b<<<dim3((C / 32) * (C2 / 16) / 8), dim3(256), 0, stream>>>(W1, W1f, C2);
  k0_pack_b<<<dim3((C / 32) * (HK / 16) / 8), dim3(256), 0, stream>>>(Ww, Wwf, HK);
  k0_pack_b<<<dim3((C / 32) * (C  / 16) / 8), dim3(256), 0, stream>>>(W2, W2f, C);
  k0_cvt_bf16<<<dim3((size_t)M * C / 8 / 256), dim3(256), 0, stream>>>(Q, Qb);

  // K1: M/16 x C/64 tiles, 8 waves/block
  k1_gemm_glu<<<dim3((M / 16) * (C / 64) / 8), dim3(256), 0, stream>>>(Qb, W1f, b1, Xb);
  // K2: one block per 16-row stripe
  k2_weights_softmax<<<dim3(M / 16), dim3(256), 0, stream>>>(Xb, Wwf, bw, Wsm);
  // K3: one thread per output element
  k3_dynconv<<<dim3((size_t)M * C / 256), dim3(256), 0, stream>>>(Xb, Wsm, mask, Yb);
  // K4: M/16 x C/64 tiles, 8 waves/block
  k4_gemm_out<<<dim3((M / 16) * (C / 64) / 8), dim3(256), 0, stream>>>(Yb, W2f, b2, Out);
}
